// TextMoE_25537875542066
// MI455X (gfx1250) — matrix operand order repaired and checked by reading, not compiled, not run on hardware
//
#include <hip/hip_runtime.h>
#include <hip/hip_bf16.h>

// ---------------------------------------------------------------------------
// TextMoE forward for MI455X (gfx1250): f16 WMMA GEMMs + flash attention.
// Dims: V=50257, E=O=128, NEXP=10, HID=256, NH=8 (hd=16), S=512, B=32, TOPK=2
// ---------------------------------------------------------------------------

typedef __attribute__((ext_vector_type(16))) _Float16 v16h;
typedef __attribute__((ext_vector_type(8)))  float    v8f;

#define TOKS   16384   // B*S
#define SEQ    512
#define BATCH  32
#define EDIM   128
#define HIDDIM 256
#define NEXPRT 10

static __device__ inline v8f v8f_splat(float x) {
  v8f r = {x, x, x, x, x, x, x, x};
  return r;
}

static __device__ inline v8f wmma16(v16h a, v16h b, v8f c) {
  // D = A(16x32 f16) * B(32x16 f16) + C(16x16 f32)
  return __builtin_amdgcn_wmma_f32_16x16x32_f16(false, a, false, b, (short)0, c,
                                                false, false);
}

// A fragment (16x32 f16), rows row0..row0+15 from a row-major f16 matrix.
// Contiguous k order (same as B): lane m = lane&15; half h -> k = (lane/16)*16 + h
static __device__ inline v16h load_a_frag(const _Float16* Abase, int lda,
                                          int row0, int kbase, int lane) {
  const _Float16* p =
      Abase + (size_t)(row0 + (lane & 15)) * lda + kbase + ((lane >> 4) & 1) * 16;
  union { uint4 u[2]; v16h h; } t;
  t.u[0] = *(const uint4*)(p);
  t.u[1] = *(const uint4*)(p + 8);
  return t.h;
}

// B fragment (32x16): B[k,n] = W[n0+n, kbase+k] with W row-major [N,K].
// Contiguous k order (as A): lanes 0-15: n=lane, k=h ; lanes 16-31: n=lane-16, k=16+h
static __device__ inline v16h load_b_frag(const _Float16* W, int ldw, int n0,
                                          int kbase, int lane) {
  const _Float16* p =
      W + (size_t)(n0 + (lane & 15)) * ldw + kbase + ((lane >> 4) & 1) * 16;
  union { uint4 u[2]; v16h h; } t;
  t.u[0] = ((const uint4*)p)[0];
  t.u[1] = ((const uint4*)p)[1];
  return t.h;
}

static __device__ inline float red_max16(float v) {
#pragma unroll
  for (int m = 8; m >= 1; m >>= 1) v = fmaxf(v, __shfl_xor(v, m, 32));
  return v;
}
static __device__ inline float red_sum16(float v) {
#pragma unroll
  for (int m = 8; m >= 1; m >>= 1) v += __shfl_xor(v, m, 32);
  return v;
}
static __device__ inline float red_sum32(float v) {
#pragma unroll
  for (int m = 16; m >= 1; m >>= 1) v += __shfl_xor(v, m, 32);
  return v;
}

// ---------------------------------------------------------------------------
// f32 -> f16 conversion
// ---------------------------------------------------------------------------
__global__ void cvt_f32_f16(const float* __restrict__ in,
                            _Float16* __restrict__ out, int n) {
  int i = blockIdx.x * 256 + threadIdx.x;
  if (i < n) out[i] = (_Float16)in[i];
}

// ---------------------------------------------------------------------------
// Embedding: x[t,c] = emb[ids[t],c] + pos[t%S, c]   (f16 out)
// ---------------------------------------------------------------------------
__global__ void embed_kernel(const float* __restrict__ emb,
                             const float* __restrict__ pos,
                             const int* __restrict__ ids,
                             _Float16* __restrict__ x16) {
  int i = blockIdx.x * 256 + threadIdx.x;   // over TOKS*EDIM
  int t = i >> 7, c = i & 127;
  int s = t & (SEQ - 1);
  int id = ids[t];
  x16[i] = (_Float16)(emb[(size_t)id * EDIM + c] + pos[s * EDIM + c]);
}

// ---------------------------------------------------------------------------
// Generic GEMM: out[T,N] = A[T,K](f16) @ W[N,K]^T(f16) + bias[N]
// 8 waves / block; block tile 32 rows x 128 cols; wave -> (rowtile, 2 coltiles)
// ---------------------------------------------------------------------------
template <int K>
__global__ __launch_bounds__(256) void gemm_bias_kernel(
    const _Float16* __restrict__ A, const _Float16* __restrict__ W,
    const float* __restrict__ bias, float* __restrict__ out32,
    _Float16* __restrict__ out16, int N) {
  __shared__ __align__(16) _Float16 As[32 * K];
  const int tid = threadIdx.x, lane = tid & 31, wave = tid >> 5;
  const int row0 = blockIdx.x * 32;
  const int ncol0 = blockIdx.y * 128;

  // prefetch this block's weight panel (128 rows x K) into cache
  if (tid < 128) __builtin_prefetch(W + (size_t)(ncol0 + tid) * K, 0, 1);

  // stage A tile (32 x K) into LDS
  const uint* Ag = (const uint*)(A + (size_t)row0 * K);
  uint* Au = (uint*)As;
  for (int i = tid; i < 32 * K / 2; i += 256) Au[i] = Ag[i];
  __syncthreads();

  const int rt = wave >> 2;        // 0..1 row tile
  const int cw = (wave & 3) * 32;  // column group
  v16h afr[K / 32];
#pragma unroll
  for (int ks = 0; ks < K / 32; ++ks)
    afr[ks] = load_a_frag(As, K, rt * 16, ks * 32, lane);

#pragma unroll
  for (int ct = 0; ct < 2; ++ct) {
    const int n0 = ncol0 + cw + ct * 16;
    v8f c = v8f_splat(bias[n0 + (lane & 15)]);
#pragma unroll
    for (int ks = 0; ks < K / 32; ++ks) {
      v16h b = load_b_frag(W, K, n0, ks * 32, lane);
      c = wmma16(afr[ks], b, c);
    }
    const int n = n0 + (lane & 15);
    const int mB = row0 + rt * 16 + ((lane >> 4) & 1) * 8;
#pragma unroll
    for (int r = 0; r < 8; ++r) {
      size_t idx = (size_t)(mB + r) * N + n;
      if (out32) out32[idx] = c[r];
      if (out16) out16[idx] = (_Float16)c[r];
    }
  }
}

// ---------------------------------------------------------------------------
// Flash attention. Block = 4 waves = 64 queries of one (b, head).
// K/V (512x16 f16) staged in LDS; head dim padded 16->32 with zeros for WMMA.
// Masking follows reference: attention_mask != 0 -> key is ignored (-1e9).
// ---------------------------------------------------------------------------

// scores-B fragment: B[k,n] = K[key0+n, k] for k<16, 0 for k>=16
static __device__ inline v16h load_kfrag(const _Float16* Ks, int key0, int lane) {
  union { uint4 u[2]; v16h h; } t;
  if (lane < 16) {
    const _Float16* p = Ks + (size_t)(key0 + lane) * 16;
    t.u[0] = ((const uint4*)p)[0];
    t.u[1] = ((const uint4*)p)[1];
  } else {
    t.u[0] = make_uint4(0u, 0u, 0u, 0u);
    t.u[1] = make_uint4(0u, 0u, 0u, 0u);
  }
  return t.h;
}

// P@V B fragment: B[k,n] = V[key0+k, n], k=(lane/16)*16+h, n=lane&15
static __device__ inline v16h load_vfrag(const _Float16* Vs, int key0, int lane) {
  v16h r;
  const _Float16* p =
      Vs + (size_t)(key0 + ((lane >> 4) & 1) * 16) * 16 + (lane & 15);
#pragma unroll
  for (int hh = 0; hh < 16; ++hh) r[hh] = p[hh * 16];
  return r;
}

__global__ __launch_bounds__(128) void attention_kernel(
    const _Float16* __restrict__ qkv, const int* __restrict__ amask,
    _Float16* __restrict__ attn_out) {
  __shared__ __align__(16) _Float16 Ks[SEQ * 16];     // 16 KB
  __shared__ __align__(16) _Float16 Vs[SEQ * 16];     // 16 KB
  __shared__ __align__(16) _Float16 Pb[4][16 * 32];   // 4 KB (per-wave P stage)

  const int tid = threadIdx.x, lane = tid & 31, wave = tid >> 5;
  const int bx = blockIdx.x;
  const int qc = bx & 7;           // query chunk (64 q)
  const int h = (bx >> 3) & 7;     // head
  const int b = bx >> 6;           // batch

  // stage K and V for this (b, head)
  for (int r = tid; r < SEQ; r += 128) {
    const _Float16* src = qkv + ((size_t)(b * SEQ + r)) * 384 + EDIM + h * 16;
    *(uint4*)(Ks + r * 16) = *(const uint4*)src;
    *(uint4*)(Ks + r * 16 + 8) = *(const uint4*)(src + 8);
    const _Float16* sv = src + EDIM;
    *(uint4*)(Vs + r * 16) = *(const uint4*)sv;
    *(uint4*)(Vs + r * 16 + 8) = *(const uint4*)(sv + 8);
  }
  __syncthreads();

  const int q0 = qc * 64 + wave * 16;
  // Q A-fragment: lanes 0-15 = q dims (k<16), lanes 16-31 = 0 (k>=16 pad)
  v16h aq;
  {
    const _Float16* qp = qkv + ((size_t)(b * SEQ + q0 + (lane & 15))) * 384 +
                         h * 16;
    union { uint4 u[2]; _Float16 hv[16]; } t;
    t.u[0] = *(const uint4*)qp;
    t.u[1] = *(const uint4*)(qp + 8);
#pragma unroll
    for (int i = 0; i < 16; ++i) {
      aq[i] = (lane < 16) ? t.hv[i] : (_Float16)0.0f;
    }
  }

  float rmax[8], rsum[8];
  v8f o = v8f_splat(0.0f);
#pragma unroll
  for (int r = 0; r < 8; ++r) { rmax[r] = -3.0e38f; rsum[r] = 0.0f; }
  const float scale = 0.25f;  // 1/sqrt(16)
  const int mrow = ((lane >> 4) & 1) * 8;

  for (int kt = 0; kt < SEQ / 32; ++kt) {
    const int kb = kt * 32;
    v16h b0 = load_kfrag(Ks, kb, lane);
    v16h b1 = load_kfrag(Ks, kb + 16, lane);
    v8f s0 = wmma16(aq, b0, v8f_splat(0.0f));
    v8f s1 = wmma16(aq, b1, v8f_splat(0.0f));
    const float m0add = (amask[b * SEQ + kb + (lane & 15)] != 0) ? -1e9f : 0.0f;
    const float m1add =
        (amask[b * SEQ + kb + 16 + (lane & 15)] != 0) ? -1e9f : 0.0f;
#pragma unroll
    for (int r = 0; r < 8; ++r) {
      float x0 = s0[r] * scale + m0add;
      float x1 = s1[r] * scale + m1add;
      float mx = red_max16(fmaxf(x0, x1));
      float nm = fmaxf(rmax[r], mx);
      float corr = __expf(rmax[r] - nm);
      float p0 = __expf(x0 - nm);
      float p1 = __expf(x1 - nm);
      rsum[r] = rsum[r] * corr + red_sum16(p0 + p1);
      rmax[r] = nm;
      o[r] *= corr;
      const int m = mrow + r;
      Pb[wave][m * 32 + (lane & 15)] = (_Float16)p0;
      Pb[wave][m * 32 + 16 + (lane & 15)] = (_Float16)p1;
    }
    // same-wave LDS ordering: DS ops are in-order, compiler inserts dscnt waits
    v16h ap = load_a_frag(Pb[wave], 32, 0, 0, lane);
    v16h bv = load_vfrag(Vs, kb, lane);
    o = wmma16(ap, bv, o);
  }

#pragma unroll
  for (int r = 0; r < 8; ++r) {
    const int m = mrow + r;
    float val = o[r] / rsum[r];
    attn_out[((size_t)(b * SEQ + q0 + m)) * EDIM + h * 16 + (lane & 15)] =
        (_Float16)val;
  }
}

// ---------------------------------------------------------------------------
// Gating: logits = a @ gw^T + gb ; softmax; top-2 renormalized -> comb[T,10]
// (softmax denominator cancels in the top-2 ratio, so use exp(l - max))
// ---------------------------------------------------------------------------
__global__ void gate_kernel(const float* __restrict__ a32,
                            const float* __restrict__ gw,
                            const float* __restrict__ gb,
                            float* __restrict__ comb) {
  int t = blockIdx.x * 256 + threadIdx.x;
  if (t >= TOKS) return;
  const float* x = a32 + (size_t)t * EDIM;
  float lg[NEXPRT];
#pragma unroll
  for (int e = 0; e < NEXPRT; ++e) {
    const float* w = gw + e * EDIM;
    float s = gb[e];
    for (int c = 0; c < EDIM; c += 4)
      s += x[c] * w[c] + x[c + 1] * w[c + 1] + x[c + 2] * w[c + 2] +
           x[c + 3] * w[c + 3];
    lg[e] = s;
  }
  float mx = lg[0];
#pragma unroll
  for (int e = 1; e < NEXPRT; ++e) mx = fmaxf(mx, lg[e]);
  float p[NEXPRT];
#pragma unroll
  for (int e = 0; e < NEXPRT; ++e) p[e] = __expf(lg[e] - mx);
  int i1 = 0;
  float v1 = p[0];
#pragma unroll
  for (int e = 1; e < NEXPRT; ++e)
    if (p[e] > v1) { v1 = p[e]; i1 = e; }
  int i2 = (i1 == 0) ? 1 : 0;
  float v2 = p[i2];
#pragma unroll
  for (int e = 0; e < NEXPRT; ++e)
    if (e != i1 && p[e] > v2) { v2 = p[e]; i2 = e; }
  float inv = 1.0f / (v1 + v2);
  float* o = comb + (size_t)t * NEXPRT;
#pragma unroll
  for (int e = 0; e < NEXPRT; ++e) o[e] = 0.0f;
  o[i1] = v1 * inv;
  o[i2] = v2 * inv;
}

// ---------------------------------------------------------------------------
// Fused MoE experts + LayerNorm + *am. Block = 32 tokens, 8 waves.
// For each expert: H = relu(A @ W1^T + b1) -> LDS ; acc += comb * (H@W2^T + b2)
// ---------------------------------------------------------------------------
__global__ __launch_bounds__(256) void moe_expert_kernel(
    const _Float16* __restrict__ a16, const float* __restrict__ comb,
    const _Float16* __restrict__ w1, const float* __restrict__ b1,
    const _Float16* __restrict__ w2, const float* __restrict__ b2,
    const float* __restrict__ lng, const float* __restrict__ lnb,
    _Float16* __restrict__ out16, float am) {
  __shared__ __align__(16) _Float16 As[32 * EDIM];    // 8 KB
  __shared__ __align__(16) _Float16 Hs[32 * HIDDIM];  // 16 KB
  __shared__ float Cs[32 * NEXPRT];                   // 1.25 KB
  __shared__ float Os[32 * EDIM];                     // 16 KB

  const int tid = threadIdx.x, lane = tid & 31, wave = tid >> 5;
  const int row0 = blockIdx.x * 32;

  {
    const uint* Ag = (const uint*)(a16 + (size_t)row0 * EDIM);
    uint* Au = (uint*)As;
    for (int i = tid; i < 32 * EDIM / 2; i += 256) Au[i] = Ag[i];
    for (int i = tid; i < 32 * NEXPRT; i += 256)
      Cs[i] = comb[(size_t)row0 * NEXPRT + i];
  }
  // prefetch expert 0 weights (64 KB each: 256 threads x 256 B)
  __builtin_prefetch(w1 + (size_t)tid * 128, 0, 1);
  __builtin_prefetch(w2 + (size_t)tid * 128, 0, 1);
  __syncthreads();

  const int rt = wave >> 2;        // 0..1
  const int cw = (wave & 3) * 32;  // output column group
  const int mrow = ((lane >> 4) & 1) * 8;
  v16h afr[4];
#pragma unroll
  for (int ks = 0; ks < 4; ++ks)
    afr[ks] = load_a_frag(As, EDIM, rt * 16, ks * 32, lane);

  v8f acc0 = v8f_splat(0.0f), acc1 = v8f_splat(0.0f);

  for (int e = 0; e < NEXPRT; ++e) {
    // prefetch next expert's weights while computing this one
    if (e + 1 < NEXPRT) {
      __builtin_prefetch(w1 + (size_t)(e + 1) * HIDDIM * EDIM +
                             (size_t)tid * 128, 0, 1);
      __builtin_prefetch(w2 + (size_t)(e + 1) * EDIM * HIDDIM +
                             (size_t)tid * 128, 0, 1);
    }
    // ---- GEMM1: H[32,256] = relu(A @ W1^T + b1) ----
    const _Float16* W1 = w1 + (size_t)e * HIDDIM * EDIM;
#pragma unroll
    for (int j = 0; j < 4; ++j) {
      const int hc = ((wave & 3) * 4 + j) * 16;
      v8f c = v8f_splat(b1[e * HIDDIM + hc + (lane & 15)]);
#pragma unroll
      for (int ks = 0; ks < 4; ++ks)
        c = wmma16(afr[ks], load_b_frag(W1, EDIM, hc, ks * 32, lane), c);
      const int nh = hc + (lane & 15);
#pragma unroll
      for (int r = 0; r < 8; ++r)
        Hs[(rt * 16 + mrow + r) * HIDDIM + nh] = (_Float16)fmaxf(c[r], 0.0f);
    }
    __syncthreads();
    // ---- GEMM2: EO[32,128] = H @ W2^T + b2, weighted accumulate ----
    // k-step outer: each A fragment is loaded once and feeds both col tiles
    const _Float16* W2 = w2 + (size_t)e * EDIM * HIDDIM;
    {
      v8f c0 = v8f_splat(b2[e * EDIM + cw + (lane & 15)]);
      v8f c1 = v8f_splat(b2[e * EDIM + cw + 16 + (lane & 15)]);
#pragma unroll
      for (int ks = 0; ks < 8; ++ks) {
        v16h ah = load_a_frag(Hs, HIDDIM, rt * 16, ks * 32, lane);
        c0 = wmma16(ah, load_b_frag(W2, HIDDIM, cw, ks * 32, lane), c0);
        c1 = wmma16(ah, load_b_frag(W2, HIDDIM, cw + 16, ks * 32, lane), c1);
      }
#pragma unroll
      for (int r = 0; r < 8; ++r) {
        float cb = Cs[(rt * 16 + mrow + r) * NEXPRT + e];
        acc0[r] += cb * c0[r];
        acc1[r] += cb * c1[r];
      }
    }
    __syncthreads();  // Hs reused next expert
  }

  // stash pre-LN sums
#pragma unroll
  for (int ct = 0; ct < 2; ++ct) {
    const v8f& acc = ct ? acc1 : acc0;
    const int n = cw + ct * 16 + (lane & 15);
#pragma unroll
    for (int r = 0; r < 8; ++r) Os[(rt * 16 + mrow + r) * EDIM + n] = acc[r];
  }
  __syncthreads();

  // LayerNorm per token (wave handles 4 tokens), then *am (= 1/S)
  for (int t = 0; t < 4; ++t) {
    const int tok = wave * 4 + t;
    float s = 0.0f, s2 = 0.0f;
    for (int c = lane; c < EDIM; c += 32) {
      float v = Os[tok * EDIM + c];
      s += v;
      s2 += v * v;
    }
    s = red_sum32(s);
    s2 = red_sum32(s2);
    float mean = s * (1.0f / EDIM);
    float var = s2 * (1.0f / EDIM) - mean * mean;
    float inv = rsqrtf(var + 1e-5f);
    for (int c = lane; c < EDIM; c += 32) {
      float v = (Os[tok * EDIM + c] - mean) * inv * lng[c] + lnb[c];
      out16[((size_t)(row0 + tok)) * EDIM + c] = (_Float16)(v * am);
    }
  }
}

// ---------------------------------------------------------------------------
// Sentence pooling: masked mean over S (a2 == 1/S), sv proj, LayerNorm
// ---------------------------------------------------------------------------
__global__ __launch_bounds__(128) void sentence_kernel(
    const float* __restrict__ sv, const int* __restrict__ amask,
    const float* __restrict__ svw, const float* __restrict__ svb,
    const float* __restrict__ lng, const float* __restrict__ lnb,
    float* __restrict__ out) {
  __shared__ float sent[EDIM];
  __shared__ float y[EDIM];
  const int b = blockIdx.x, c = threadIdx.x;
  float acc = 0.0f, msum = 0.0f;
  for (int s = 0; s < SEQ; ++s) {
    float mv = (float)amask[b * SEQ + s];
    acc += sv[((size_t)(b * SEQ + s)) * EDIM + c] * mv;
    msum += mv;
  }
  const float am = 1.0f / (float)SEQ;
  sent[c] = acc * am / msum;
  __syncthreads();
  float s2 = svb[c];
  const float* w = svw + c * EDIM;
  for (int k = 0; k < EDIM; ++k) s2 += sent[k] * w[k];
  y[c] = s2;
  __syncthreads();
  float mean = 0.0f;
  for (int k = 0; k < EDIM; ++k) mean += y[k];
  mean *= (1.0f / EDIM);
  float var = 0.0f;
  for (int k = 0; k < EDIM; ++k) {
    float d = y[k] - mean;
    var += d * d;
  }
  var *= (1.0f / EDIM);
  out[b * EDIM + c] = (y[c] - mean) * rsqrtf(var + 1e-5f) * lng[c] + lnb[c];
}

// ---------------------------------------------------------------------------
// Host launcher
// ---------------------------------------------------------------------------
extern "C" void kernel_launch(void* const* d_in, const int* in_sizes, int n_in,
                              void* d_out, int out_size, void* d_ws,
                              size_t ws_size, hipStream_t stream) {
  (void)in_sizes; (void)n_in; (void)out_size; (void)ws_size;
  const float* emb    = (const float*)d_in[0];
  const float* pos    = (const float*)d_in[1];
  const float* proj_w = (const float*)d_in[2];
  const float* proj_b = (const float*)d_in[3];
  const float* qkvw   = (const float*)d_in[4];
  const float* qkvb   = (const float*)d_in[5];
  const float* ow     = (const float*)d_in[6];
  const float* ob     = (const float*)d_in[7];
  const float* gw     = (const float*)d_in[8];
  const float* gb     = (const float*)d_in[9];
  const float* ew1    = (const float*)d_in[10];
  const float* eb1    = (const float*)d_in[11];
  const float* ew2    = (const float*)d_in[12];
  const float* eb2    = (const float*)d_in[13];
  const float* mlng   = (const float*)d_in[14];
  const float* mlnb   = (const float*)d_in[15];
  const float* vecw   = (const float*)d_in[16];
  const float* vecb   = (const float*)d_in[17];
  const float* svw    = (const float*)d_in[18];
  const float* svb    = (const float*)d_in[19];
  const float* lng    = (const float*)d_in[20];
  const float* lnb    = (const float*)d_in[21];
  const int*   ids    = (const int*)d_in[22];
  const int*   amask  = (const int*)d_in[23];

  char* ws = (char*)d_ws;
  size_t off = 0;
  auto carve = [&](size_t bytes) -> char* {
    char* p = ws + off;
    off += (bytes + 255) & ~(size_t)255;
    return p;
  };

  _Float16* x16    = (_Float16*)carve((size_t)TOKS * EDIM * 2);
  _Float16* y16    = (_Float16*)carve((size_t)TOKS * EDIM * 2);
  _Float16* qkv16  = (_Float16*)carve((size_t)TOKS * 384 * 2);
  _Float16* attn16 = (_Float16*)carve((size_t)TOKS * EDIM * 2);
  float*    a32    = (float*)carve((size_t)TOKS * EDIM * 4);
  _Float16* a16    = (_Float16*)carve((size_t)TOKS * EDIM * 2);
  float*    comb   = (float*)carve((size_t)TOKS * NEXPRT * 4);
  _Float16* moe16  = (_Float16*)carve((size_t)TOKS * EDIM * 2);
  _Float16* fv16   = (_Float16*)carve((size_t)TOKS * EDIM * 2);
  _Float16* pw16   = (_Float16*)carve((size_t)2 * EDIM * EDIM * 2);
  _Float16* qw16   = (_Float16*)carve((size_t)2 * 384 * EDIM * 2);
  _Float16* ow16   = (_Float16*)carve((size_t)2 * EDIM * EDIM * 2);
  _Float16* w116   = (_Float16*)carve((size_t)2 * NEXPRT * HIDDIM * EDIM * 2);
  _Float16* w216   = (_Float16*)carve((size_t)2 * NEXPRT * EDIM * HIDDIM * 2);
  _Float16* vw16   = (_Float16*)carve((size_t)EDIM * EDIM * 2);

  auto cvt = [&](const float* src, _Float16* dst, int n) {
    cvt_f32_f16<<<(n + 255) / 256, 256, 0, stream>>>(src, dst, n);
  };
  cvt(proj_w, pw16, 2 * EDIM * EDIM);
  cvt(qkvw, qw16, 2 * 384 * EDIM);
  cvt(ow, ow16, 2 * EDIM * EDIM);
  cvt(ew1, w116, 2 * NEXPRT * HIDDIM * EDIM);
  cvt(ew2, w216, 2 * NEXPRT * EDIM * HIDDIM);
  cvt(vecw, vw16, EDIM * EDIM);

  embed_kernel<<<(TOKS * EDIM) / 256, 256, 0, stream>>>(emb, pos, ids, x16);

  const float am = 1.0f / (float)SEQ;  // softmax rows sum to 1 -> aw.mean == 1/S
  float* fv32 = (float*)d_out;                           // first_vector
  float* sv32 = (float*)d_out + (size_t)TOKS * EDIM;     // second_vector
  float* sent = (float*)d_out + (size_t)2 * TOKS * EDIM; // sentence

  for (int L = 0; L < 2; ++L) {
    const _Float16* lin = (L == 0) ? x16 : fv16;
    // input projection
    gemm_bias_kernel<EDIM><<<dim3(TOKS / 32, 1), 256, 0, stream>>>(
        lin, pw16 + (size_t)L * EDIM * EDIM, proj_b + L * EDIM, nullptr, y16,
        EDIM);
    // qkv
    gemm_bias_kernel<EDIM><<<dim3(TOKS / 32, 3), 256, 0, stream>>>(
        y16, qw16 + (size_t)L * 384 * EDIM, qkvb + L * 384, nullptr, qkv16,
        384);
    // attention (flash, per (b, head, 64-query chunk))
    attention_kernel<<<BATCH * 8 * (SEQ / 64), 128, 0, stream>>>(qkv16, amask,
                                                                 attn16);
    // output projection (f32 for gating, f16 for experts)
    gemm_bias_kernel<EDIM><<<dim3(TOKS / 32, 1), 256, 0, stream>>>(
        attn16, ow16 + (size_t)L * EDIM * EDIM, ob + L * EDIM, a32, a16, EDIM);
    // gating -> comb
    gate_kernel<<<TOKS / 256, 256, 0, stream>>>(a32, gw + L * NEXPRT * EDIM,
                                                gb + L * NEXPRT, comb);
    // fused experts + LN + *am
    moe_expert_kernel<<<TOKS / 32, 256, 0, stream>>>(
        a16, comb, w116 + (size_t)L * NEXPRT * HIDDIM * EDIM,
        eb1 + L * NEXPRT * HIDDIM, w216 + (size_t)L * NEXPRT * EDIM * HIDDIM,
        eb2 + L * NEXPRT * EDIM, mlng + L * EDIM, mlnb + L * EDIM, moe16, am);
    // vector projection -> d_out region (+ f16 copy feeding layer 2)
    gemm_bias_kernel<EDIM><<<dim3(TOKS / 32, 1), 256, 0, stream>>>(
        moe16, vw16, vecb, (L == 0) ? fv32 : sv32, (L == 0) ? fv16 : nullptr,
        EDIM);
  }

  sentence_kernel<<<BATCH, 128, 0, stream>>>(sv32, amask, svw, svb, lng, lnb,
                                             sent);
}
